// DependencyParser_85830626443417
// MI455X (gfx1250) — matrix operand order repaired and checked by reading, not compiled, not run on hardware
//
#include <hip/hip_runtime.h>
#include <hip/hip_bf16.h>

typedef __attribute__((ext_vector_type(16))) __bf16 bf16x16;
typedef __attribute__((ext_vector_type(8)))  __bf16 bf16x8;
typedef __attribute__((ext_vector_type(8)))  float  f32x8;

// Fast transcendental helpers (native gfx1250 ops when the builtin exists).
#if __has_builtin(__builtin_amdgcn_tanhf)
__device__ __forceinline__ float fast_tanh(float x) { return __builtin_amdgcn_tanhf(x); }
#elif __has_builtin(__builtin_amdgcn_tanh_f32)
__device__ __forceinline__ float fast_tanh(float x) { return __builtin_amdgcn_tanh_f32(x); }
#else
__device__ __forceinline__ float fast_tanh(float x) { return tanhf(x); }
#endif

#if __has_builtin(__builtin_amdgcn_rcpf)
__device__ __forceinline__ float fast_rcp(float x) { return __builtin_amdgcn_rcpf(x); }
#else
__device__ __forceinline__ float fast_rcp(float x) { return 1.0f / x; }
#endif

__device__ __forceinline__ float fast_sigmoid(float x) {
    return fast_rcp(1.0f + __expf(-x));
}

// ---------------------------------------------------------------------------
// Pack an f32 weight [Ntrue x Ktrue] into zero-padded bf16 [Npad x Kpad]
// ---------------------------------------------------------------------------
__global__ void pack_bf16(const float* __restrict__ src, __bf16* __restrict__ dst,
                          int Ntrue, int Ktrue, int Kpad, int total) {
    int i = blockIdx.x * blockDim.x + threadIdx.x;
    if (i >= total) return;
    int n = i / Kpad, k = i % Kpad;
    float v = (n < Ntrue && k < Ktrue) ? src[n * Ktrue + k] : 0.0f;
    dst[i] = (__bf16)v;
}

// ---------------------------------------------------------------------------
// Embedding gather: X0[t][0:100]=word_emb[widx[t]], [100:125]=pos_emb, pad->128
// ---------------------------------------------------------------------------
__global__ __launch_bounds__(128)
void embed_kernel(const int* __restrict__ widx, const int* __restrict__ pidx,
                  const float* __restrict__ wemb, const float* __restrict__ pemb,
                  __bf16* __restrict__ X0) {
    int t = blockIdx.x;
    int j = threadIdx.x;
    float v = 0.0f;
    if (j < 100)       v = wemb[(size_t)widx[t] * 100 + j];
    else if (j < 125)  v = pemb[(size_t)pidx[t] * 25 + (j - 100)];
    X0[(size_t)t * 128 + j] = (__bf16)v;
}

// ---------------------------------------------------------------------------
// Generic bf16 WMMA GEMM: C[M x Npad] = A[M x Kpad] * B^T  (B stored [Npad][Kpad])
// One wave computes one 16x16 tile. bias1/bias2 (f32, len Ntrue) optional.
// ---------------------------------------------------------------------------
__global__ __launch_bounds__(32)
void gemm_bf16_wmma(const __bf16* __restrict__ A, const __bf16* __restrict__ B,
                    const float* __restrict__ bias1, const float* __restrict__ bias2,
                    float* __restrict__ C, int Npad, int Ntrue, int Kpad) {
    const int ntiles = Npad >> 4;
    const int tileM  = blockIdx.x / ntiles;
    const int tileN  = blockIdx.x % ntiles;
    const int lane   = threadIdx.x;
    const int l16    = lane & 15;
    const int hi     = lane >> 4;
    const int rowA   = tileM * 16 + l16;
    const int colB   = tileN * 16 + l16;

    const __bf16* Arow = A + (size_t)rowA * Kpad;
    const __bf16* Brow = B + (size_t)colB * Kpad;

    f32x8 acc = {};
    for (int kb = 0; kb < Kpad; kb += 32) {
        // A fragment: lane-half hi holds K = kb+8*hi..+7 (elems 0-7) and
        //             K = kb+16+8*hi..+7 (elems 8-15)   [ISA 16-bit A layout]
        bf16x8 alo = *(const bf16x8*)(Arow + kb + hi * 8);
        bf16x8 ahi = *(const bf16x8*)(Arow + kb + 16 + hi * 8);
        bf16x16 a, b;
        #pragma unroll
        for (int e = 0; e < 8; ++e) { a[e] = alo[e]; a[e + 8] = ahi[e]; }
        // B fragment: lane = column, same K layout as A
        bf16x8 blo = *(const bf16x8*)(Brow + kb + hi * 8);
        bf16x8 bhi = *(const bf16x8*)(Brow + kb + 16 + hi * 8);
        #pragma unroll
        for (int e = 0; e < 8; ++e) { b[e] = blo[e]; b[e + 8] = bhi[e]; }
        acc = __builtin_amdgcn_wmma_f32_16x16x32_bf16(false, a, false, b,
                                                      (short)0, acc, false, false);
    }
    float bv = 0.0f;
    if (colB < Ntrue) {
        if (bias1) bv += bias1[colB];
        if (bias2) bv += bias2[colB];
    }
    #pragma unroll
    for (int j = 0; j < 8; ++j) {
        int r = tileM * 16 + j + hi * 8;   // D layout: VGPR j -> row j + 8*hi
        C[(size_t)r * Npad + colB] = acc[j] + bv;
    }
}

// ---------------------------------------------------------------------------
// LSTM scan (one block per direction). Recurrent h@Whh^T via WMMA: h is
// broadcast into ALL 16 rows of the A fragment (every lane reads identical
// LDS addresses -> bank broadcast, no masking VALU); D row 0 is consumed.
// Whh bf16 [512 x 128]. G = precomputed x@Wih^T + b (f32, [1024 x 512]).
// Writes h (bf16) into Xnext[1024 x 256] at column offset 0 (fwd) / 125 (bwd).
// ---------------------------------------------------------------------------
__global__ __launch_bounds__(128)
void lstm_scan(const float* __restrict__ Gf, const float* __restrict__ Gb,
               const __bf16* __restrict__ WhhF, const __bf16* __restrict__ WhhB,
               __bf16* __restrict__ Xnext) {
    __shared__ __bf16 h_bf[128];
    __shared__ float  gates[512];

    const int tid  = threadIdx.x;
    const int lane = tid & 31;
    const int wave = tid >> 5;
    const int l16  = lane & 15;
    const int hi   = lane >> 4;

    const int rev = blockIdx.x;            // 0 = forward, 1 = backward
    const float*  G = rev ? Gb : Gf;
    const __bf16* W = rev ? WhhB : WhhF;
    const int xoff  = rev ? 125 : 0;

    if (tid < 128) h_bf[tid] = (__bf16)0.0f;
    float c = 0.0f, h = 0.0f;
    __syncthreads();

    for (int t = 0; t < 1024; ++t) {
        const int idx = rev ? (1023 - t) : t;
        // gates_rec[n] = sum_k h[k] * Whh[n][k], n in [0,512)
        #pragma unroll
        for (int nt = 0; nt < 8; ++nt) {
            const int colB = (wave * 8 + nt) * 16 + l16;
            const __bf16* Brow = W + (size_t)colB * 128;
            f32x8 acc = {};
            #pragma unroll
            for (int kb = 0; kb < 128; kb += 32) {
                // Broadcast h into every A row: address depends only on hi.
                bf16x8 alo = *(const bf16x8*)(h_bf + kb + hi * 8);
                bf16x8 ahi = *(const bf16x8*)(h_bf + kb + 16 + hi * 8);
                bf16x16 a, b;
                #pragma unroll
                for (int e = 0; e < 8; ++e) { a[e] = alo[e]; a[e + 8] = ahi[e]; }
                bf16x8 blo = *(const bf16x8*)(Brow + kb + hi * 8);
                bf16x8 bhi = *(const bf16x8*)(Brow + kb + 16 + hi * 8);
                #pragma unroll
                for (int e = 0; e < 8; ++e) { b[e] = blo[e]; b[e + 8] = bhi[e]; }
                acc = __builtin_amdgcn_wmma_f32_16x16x32_bf16(false, a, false, b,
                                                              (short)0, acc, false, false);
            }
            if (hi == 0)   // row M=0 lives in acc[0] of lanes 0..15
                gates[colB] = acc[0] + G[(size_t)idx * 512 + colB];
        }
        __syncthreads();
        if (tid < 125) {
            float iv = gates[tid];
            float fv = gates[125 + tid];
            float gv = gates[250 + tid];
            float ov = gates[375 + tid];
            float si = fast_sigmoid(iv);
            float sf = fast_sigmoid(fv);
            float so = fast_sigmoid(ov);
            c = sf * c + si * fast_tanh(gv);
            h = so * fast_tanh(c);
            h_bf[tid] = (__bf16)h;
            Xnext[(size_t)idx * 256 + xoff + tid] = (__bf16)h;
        }
        __syncthreads();
    }
}

// ---------------------------------------------------------------------------
// Biaffine scoring: out[h][m] = b3 + sum_d w3[d]*tanh(heads[h][d]+mods[m][d]),
// diagonal = -inf. heads/mods are f32 [1024 x 112] (100 valid cols).
// ---------------------------------------------------------------------------
__global__ __launch_bounds__(256)
void score_kernel(const float* __restrict__ heads, const float* __restrict__ mods,
                  const float* __restrict__ w3, const float* __restrict__ b3,
                  float* __restrict__ out) {
    __shared__ float shH[16][100];
    __shared__ float shM[16][100];
    __shared__ float shW[100];
    const int tx = threadIdx.x, ty = threadIdx.y;
    const int tid = ty * 16 + tx;
    const int h0 = blockIdx.y * 16, m0 = blockIdx.x * 16;
    for (int i = tid; i < 1600; i += 256) {
        int r = i / 100, d = i % 100;
        shH[r][d] = heads[(size_t)(h0 + r) * 112 + d];
        shM[r][d] = mods [(size_t)(m0 + r) * 112 + d];
    }
    if (tid < 100) shW[tid] = w3[tid];
    __syncthreads();
    const int hrow = h0 + ty, mcol = m0 + tx;
    float s = b3[0];
    #pragma unroll 5
    for (int d = 0; d < 100; ++d)
        s += shW[d] * fast_tanh(shH[ty][d] + shM[tx][d]);
    out[(size_t)hrow * 1024 + mcol] = (hrow == mcol) ? -__builtin_inff() : s;
}

// ---------------------------------------------------------------------------
extern "C" void kernel_launch(void* const* d_in, const int* in_sizes, int n_in,
                              void* d_out, int out_size, void* d_ws, size_t ws_size,
                              hipStream_t stream) {
    (void)in_sizes; (void)n_in; (void)out_size; (void)ws_size;

    const int*   widx = (const int*)d_in[0];
    const int*   pidx = (const int*)d_in[1];
    const float* wemb = (const float*)d_in[3];
    const float* pemb = (const float*)d_in[4];
    // LSTM params: (w_ih, w_hh, b_ih, b_hh) x {0f,0b,1f,1b} at 5..20
    const float* w1 = (const float*)d_in[21];
    const float* b1 = (const float*)d_in[22];
    const float* w2 = (const float*)d_in[23];
    const float* b2 = (const float*)d_in[24];
    const float* w3 = (const float*)d_in[25];
    const float* b3 = (const float*)d_in[26];
    float* out = (float*)d_out;

    // ---------------- workspace layout (bytes) ----------------
    char* ws = (char*)d_ws;
    size_t off = 0;
    auto alloc = [&](size_t bytes) { char* p = ws + off; off += (bytes + 255) & ~(size_t)255; return p; };
    __bf16* X0    = (__bf16*)alloc(1024 * 128 * 2);   // layer-0 input (bf16, Kpad=128)
    __bf16* X1    = (__bf16*)alloc(1024 * 256 * 2);   // layer-1 input (bf16, Kpad=256)
    __bf16* H1    = (__bf16*)alloc(1024 * 256 * 2);   // layer-1 output (bf16)
    __bf16* Wih0f = (__bf16*)alloc(512 * 128 * 2);
    __bf16* Wih0b = (__bf16*)alloc(512 * 128 * 2);
    __bf16* Whh0f = (__bf16*)alloc(512 * 128 * 2);
    __bf16* Whh0b = (__bf16*)alloc(512 * 128 * 2);
    __bf16* Wih1f = (__bf16*)alloc(512 * 256 * 2);
    __bf16* Wih1b = (__bf16*)alloc(512 * 256 * 2);
    __bf16* Whh1f = (__bf16*)alloc(512 * 128 * 2);
    __bf16* Whh1b = (__bf16*)alloc(512 * 128 * 2);
    __bf16* W1b   = (__bf16*)alloc(112 * 256 * 2);
    __bf16* W2b   = (__bf16*)alloc(112 * 256 * 2);
    float*  Gf    = (float*)alloc(1024 * 512 * 4);    // gate pre-activations fwd
    float*  Gb    = (float*)alloc(1024 * 512 * 4);    // gate pre-activations bwd
    float*  headsF= (float*)alloc(1024 * 112 * 4);
    float*  modsF = (float*)alloc(1024 * 112 * 4);

    // ---------------- weight packing ----------------
    auto pack = [&](const void* src, __bf16* dst, int Nt, int Kt, int Kp, int Np) {
        int total = Np * Kp;
        pack_bf16<<<(total + 255) / 256, 256, 0, stream>>>((const float*)src, dst, Nt, Kt, Kp, total);
    };
    pack(d_in[5],  Wih0f, 500, 125, 128, 512);
    pack(d_in[6],  Whh0f, 500, 125, 128, 512);
    pack(d_in[9],  Wih0b, 500, 125, 128, 512);
    pack(d_in[10], Whh0b, 500, 125, 128, 512);
    pack(d_in[13], Wih1f, 500, 250, 256, 512);
    pack(d_in[14], Whh1f, 500, 125, 128, 512);
    pack(d_in[17], Wih1b, 500, 250, 256, 512);
    pack(d_in[18], Whh1b, 500, 125, 128, 512);
    pack(w1,       W1b,   100, 250, 256, 112);
    pack(w2,       W2b,   100, 250, 256, 112);

    // zero pad columns of the concat buffers (scan only writes 2x125 cols)
    hipMemsetAsync(X1, 0, 1024 * 256 * 2, stream);
    hipMemsetAsync(H1, 0, 1024 * 256 * 2, stream);

    // ---------------- embeddings ----------------
    embed_kernel<<<1024, 128, 0, stream>>>(widx, pidx, wemb, pemb, X0);

    // ---------------- layer 0 ----------------
    gemm_bf16_wmma<<<64 * 32, 32, 0, stream>>>(X0, Wih0f, (const float*)d_in[7],
                                               (const float*)d_in[8],  Gf, 512, 500, 128);
    gemm_bf16_wmma<<<64 * 32, 32, 0, stream>>>(X0, Wih0b, (const float*)d_in[11],
                                               (const float*)d_in[12], Gb, 512, 500, 128);
    lstm_scan<<<2, 128, 0, stream>>>(Gf, Gb, Whh0f, Whh0b, X1);

    // ---------------- layer 1 ----------------
    gemm_bf16_wmma<<<64 * 32, 32, 0, stream>>>(X1, Wih1f, (const float*)d_in[15],
                                               (const float*)d_in[16], Gf, 512, 500, 256);
    gemm_bf16_wmma<<<64 * 32, 32, 0, stream>>>(X1, Wih1b, (const float*)d_in[19],
                                               (const float*)d_in[20], Gb, 512, 500, 256);
    lstm_scan<<<2, 128, 0, stream>>>(Gf, Gb, Whh1f, Whh1b, H1);

    // ---------------- projections ----------------
    gemm_bf16_wmma<<<64 * 7, 32, 0, stream>>>(H1, W1b, b1, nullptr, headsF, 112, 100, 256);
    gemm_bf16_wmma<<<64 * 7, 32, 0, stream>>>(H1, W2b, b2, nullptr, modsF,  112, 100, 256);

    // ---------------- biaffine scoring ----------------
    score_kernel<<<dim3(64, 64), dim3(16, 16), 0, stream>>>(headsF, modsF, w3, b3, out);
}
